// RelativeSelfMultiheadAttn_86406152061740
// MI455X (gfx1250) — matrix operand order repaired and checked by reading, not compiled, not run on hardware
//
#include <hip/hip_runtime.h>

#define T_DIM 1024
#define B_DIM 2
#define E_DIM 1024
#define H_DIM 16
#define D_DIM 64
#define R_PAD 2048
#define MROWS (T_DIM * B_DIM) /* 2048 */

typedef __bf16 bf16_t;
typedef __attribute__((ext_vector_type(16))) __bf16 v16bf;
typedef __attribute__((ext_vector_type(8)))  __bf16 v8bf;
typedef __attribute__((ext_vector_type(8)))  float  v8f;

union V16U { v16bf v; v8bf h[2]; };

static __device__ __forceinline__ v8f wmma_bf16(v16bf a, v16bf b, v8f c) {
  // D = A(16x32) * B(32x16) + C, f32 accumulate
  return __builtin_amdgcn_wmma_f32_16x16x32_bf16(false, a, false, b, (short)0, c,
                                                 false, false);
}

// Async global->LDS copy (CDNA5, tracked by ASYNCcnt). lds_off = LDS byte addr.
static __device__ __forceinline__ void async_load_b128(unsigned int lds_off,
                                                       const bf16_t* gaddr) {
  asm volatile("global_load_async_to_lds_b128 %0, %1, off"
               :: "v"(lds_off), "v"(gaddr)
               : "memory");
}

static __device__ __forceinline__ void wait_async0() {
  asm volatile("s_wait_asynccnt 0" ::: "memory");
}

// ---------------- elementwise converters ----------------
__global__ void cvt_f32_bf16(const float* __restrict__ s, bf16_t* __restrict__ d, int n) {
  for (int i = blockIdx.x * blockDim.x + threadIdx.x; i < n; i += gridDim.x * blockDim.x)
    d[i] = (bf16_t)s[i];
}

__global__ void cvt_pad_rows(const float* __restrict__ s, bf16_t* __restrict__ d,
                             int valid_rows, int rowlen, int n) {
  for (int i = blockIdx.x * blockDim.x + threadIdx.x; i < n; i += gridDim.x * blockDim.x) {
    int r = i / rowlen;
    d[i] = (r < valid_rows) ? (bf16_t)s[i] : (bf16_t)0.0f;
  }
}

// ---------------- generic WMMA GEMM: C[M,N] = A[M,K] * B[N,K]^T + bias[N] ----------------
__global__ __launch_bounds__(128)
void gemm_bf16_wmma(float* __restrict__ C, const bf16_t* __restrict__ A,
                    const bf16_t* __restrict__ Bw, const float* __restrict__ bias,
                    int M, int N, int K) {
  const int lane = threadIdx.x & 31;
  const int w    = threadIdx.x >> 5;
  const int g    = lane >> 4;   // 16-lane half
  const int ln   = lane & 15;
  const int n0   = blockIdx.x * 64;
  const int m0   = blockIdx.y * 64 + w * 16;

  const bf16_t* aRow = A + (size_t)(m0 + ln) * K;

  v8f zero = {0.f, 0.f, 0.f, 0.f, 0.f, 0.f, 0.f, 0.f};
  v8f acc[4];
#pragma unroll
  for (int t = 0; t < 4; ++t) acc[t] = zero;

  for (int k0 = 0; k0 < K; k0 += 32) {
    // A operand: row m = ln, K = g*16..+15
    V16U a;
    a.h[0] = *(const v8bf*)(aRow + k0 + g * 16);
    a.h[1] = *(const v8bf*)(aRow + k0 + g * 16 + 8);
#pragma unroll
    for (int nt = 0; nt < 4; ++nt) {
      // B operand: column n = ln, K = g*16..+15 -> contiguous run of Bw row
      const bf16_t* bRow = Bw + (size_t)(n0 + nt * 16 + ln) * K + k0 + g * 16;
      acc[nt] = wmma_bf16(a.v, *(const v16bf*)bRow, acc[nt]);
    }
  }
#pragma unroll
  for (int nt = 0; nt < 4; ++nt) {
    float bv = bias[n0 + nt * 16 + ln];
#pragma unroll
    for (int i = 0; i < 8; ++i) {
      C[(size_t)(m0 + i + 8 * g) * N + (n0 + nt * 16 + ln)] = acc[nt][i] + bv;
    }
  }
}

// ---------------- split / bias / transpose qkv ----------------
__global__ void prep_qkv(const float* __restrict__ qkv, const float* __restrict__ rwb,
                         const float* __restrict__ rrb, bf16_t* __restrict__ qw,
                         bf16_t* __restrict__ qr, bf16_t* __restrict__ kk,
                         bf16_t* __restrict__ vT) {
  const int n = B_DIM * H_DIM * T_DIM * D_DIM;
  for (int i = blockIdx.x * blockDim.x + threadIdx.x; i < n; i += gridDim.x * blockDim.x) {
    int d = i & (D_DIM - 1);
    int t = (i >> 6) & (T_DIM - 1);
    int h = (i >> 16) & (H_DIM - 1);
    int b = i >> 20;
    int row = t * B_DIM + b;
    int col = h * D_DIM + d;
    float qv = qkv[(size_t)row * (3 * E_DIM) + col];
    float kv = qkv[(size_t)row * (3 * E_DIM) + E_DIM + col];
    float vv = qkv[(size_t)row * (3 * E_DIM) + 2 * E_DIM + col];
    size_t o = ((size_t)(b * H_DIM + h) * T_DIM + t) * D_DIM + d;
    qw[o] = (bf16_t)(qv + rwb[col]);
    qr[o] = (bf16_t)(qv + rrb[col]);
    kk[o] = (bf16_t)kv;
    vT[((size_t)(b * H_DIM + h) * D_DIM + d) * T_DIM + t] = (bf16_t)vv;
  }
}

// ---------------- fused rel-attention (flash-style, WMMA + async-LDS staging) ----------------
__global__ __launch_bounds__(128)
void attn_wmma(const bf16_t* __restrict__ qw, const bf16_t* __restrict__ qr,
               const bf16_t* __restrict__ kmat, const bf16_t* __restrict__ vT,
               const bf16_t* __restrict__ pmat, bf16_t* __restrict__ ctx) {
  __shared__ bf16_t kTile[2][32][64];  // [buf][s_rel][d]      4KB each
  __shared__ bf16_t vTile[2][64][32];  // [buf][d][s_rel]      4KB each
  __shared__ bf16_t pA[4][16][32];     // per-wave prob tile (A layout staging)

  const int tid  = threadIdx.x;
  const int lane = tid & 31;
  const int w    = tid >> 5;
  const int g    = lane >> 4;
  const int ln   = lane & 15;
  const int bh   = blockIdx.y;
  const int b    = bh >> 4;
  const int h    = bh & 15;
  const int t0   = blockIdx.x * 64 + w * 16;

  const bf16_t* kBase = kmat + (size_t)bh * T_DIM * D_DIM;
  const bf16_t* vBase = vT + (size_t)bh * D_DIM * T_DIM;
  const bf16_t* pCol  = pmat + h * D_DIM;

  const unsigned int kLds = (unsigned int)(uintptr_t)&kTile[0][0][0];
  const unsigned int vLds = (unsigned int)(uintptr_t)&vTile[0][0][0];

  // Query operands (resident whole kernel): K=64 -> two 16x32 A operands each
  const bf16_t* qwRow = qw + ((size_t)bh * T_DIM + t0 + ln) * D_DIM;
  const bf16_t* qrRow = qr + ((size_t)bh * T_DIM + t0 + ln) * D_DIM;
  V16U aqw0, aqw1, aqr0, aqr1;
  aqw0.h[0] = *(const v8bf*)(qwRow + g * 16);
  aqw0.h[1] = *(const v8bf*)(qwRow + g * 16 + 8);
  aqw1.h[0] = *(const v8bf*)(qwRow + 32 + g * 16);
  aqw1.h[1] = *(const v8bf*)(qwRow + 32 + g * 16 + 8);
  aqr0.h[0] = *(const v8bf*)(qrRow + g * 16);
  aqr0.h[1] = *(const v8bf*)(qrRow + g * 16 + 8);
  aqr1.h[0] = *(const v8bf*)(qrRow + 32 + g * 16);
  aqr1.h[1] = *(const v8bf*)(qrRow + 32 + g * 16 + 8);

  v8f zero = {0.f, 0.f, 0.f, 0.f, 0.f, 0.f, 0.f, 0.f};
  v8f ctxAcc[4];
#pragma unroll
  for (int t = 0; t < 4; ++t) ctxAcc[t] = zero;
  float mrow[8], lrow[8];
#pragma unroll
  for (int i = 0; i < 8; ++i) { mrow[i] = -3.0e38f; lrow[i] = 0.f; }

  // Issue async staging of one 32-key tile (K: contiguous 4KB; V^T: 64 strided rows).
  auto issue_tiles = [&](int buf, int s0n) {
    const bf16_t* kg = kBase + (size_t)s0n * D_DIM;
    unsigned int kb = kLds + (unsigned int)buf * 4096u;
    async_load_b128(kb + (unsigned int)tid * 16u, kg + tid * 8);
    async_load_b128(kb + (unsigned int)(tid + 128) * 16u, kg + (tid + 128) * 8);
    unsigned int vb = vLds + (unsigned int)buf * 4096u;
    int d0 = tid >> 2, c0 = tid & 3;
    async_load_b128(vb + (unsigned int)(d0 * 32 + c0 * 8) * 2u,
                    vBase + (size_t)d0 * T_DIM + s0n + c0 * 8);
    int i2 = tid + 128;
    int d1 = i2 >> 2, c1 = i2 & 3;
    async_load_b128(vb + (unsigned int)(d1 * 32 + c1 * 8) * 2u,
                    vBase + (size_t)d1 * T_DIM + s0n + c1 * 8);
  };

  issue_tiles(0, 0);

  for (int s0 = 0; s0 < T_DIM; s0 += 32) {
    const int buf = (s0 >> 5) & 1;
    wait_async0();     // current tile landed in LDS (issued last iteration)
    __syncthreads();   // visible to all waves; everyone done reading buf^1
    if (s0 + 32 < T_DIM) issue_tiles(buf ^ 1, s0 + 32);  // overlap next copy

    // ---- content scores AC = (q+r_w) . k^T  (two 16-wide key subtiles) ----
    v8f ac[2];
    ac[0] = zero; ac[1] = zero;
#pragma unroll
    for (int j = 0; j < 2; ++j) {
      const bf16_t* kr = &kTile[buf][16 * j + ln][g * 16];
      ac[j] = wmma_bf16(aqw0.v, *(const v16bf*)kr, ac[j]);
      ac[j] = wmma_bf16(aqw1.v, *(const v16bf*)(kr + 32), ac[j]);
    }

    // ---- position scores: band GEMM E = (q+r_r) . p_band^T, then diagonal gather ----
    float sc[2][8];
#pragma unroll
    for (int j = 0; j < 2; ++j) {
      int rb = (T_DIM - 1) + s0 + 16 * j - t0;  // p row for (t0, s_subtile_base)
      const bf16_t* p0r = pCol + (size_t)(rb - 15 + ln) * E_DIM + g * 16;  // c = 0..15
      const bf16_t* p1r = pCol + (size_t)(rb + 1 + ln) * E_DIM + g * 16;   // c = 16..31
      v8f e0 = zero, e1 = zero;
      e0 = wmma_bf16(aqr0.v, *(const v16bf*)p0r, e0);
      e0 = wmma_bf16(aqr1.v, *(const v16bf*)(p0r + 32), e0);
      e1 = wmma_bf16(aqr0.v, *(const v16bf*)p1r, e1);
      e1 = wmma_bf16(aqr1.v, *(const v16bf*)(p1r + 32), e1);
#pragma unroll
      for (int i = 0; i < 8; ++i) {
        int m   = i + 8 * g;
        int c   = ln - m + 15;            // rel-shift diagonal index, 0..30
        int src = (c & 15) | (g << 4);    // same half-wave holds row m
        float g0 = __shfl(e0[i], src, 32);
        float g1 = __shfl(e1[i], src, 32);
        float bd = (c < 16) ? g0 : g1;
        sc[j][i] = (ac[j][i] + bd) * 0.125f;  // 1/sqrt(64)
      }
    }

    // ---- online softmax (rows live in 16-lane halves) ----
#pragma unroll
    for (int i = 0; i < 8; ++i) {
      float loc = fmaxf(sc[0][i], sc[1][i]);
      loc = fmaxf(loc, __shfl_xor(loc, 1, 32));
      loc = fmaxf(loc, __shfl_xor(loc, 2, 32));
      loc = fmaxf(loc, __shfl_xor(loc, 4, 32));
      loc = fmaxf(loc, __shfl_xor(loc, 8, 32));
      float mnew  = fmaxf(mrow[i], loc);
      float alpha = __expf(mrow[i] - mnew);
      float p0    = __expf(sc[0][i] - mnew);
      float p1    = __expf(sc[1][i] - mnew);
      mrow[i] = mnew;
      float ps = p0 + p1;
      ps += __shfl_xor(ps, 1, 32);
      ps += __shfl_xor(ps, 2, 32);
      ps += __shfl_xor(ps, 4, 32);
      ps += __shfl_xor(ps, 8, 32);
      lrow[i] = lrow[i] * alpha + ps;
#pragma unroll
      for (int dt = 0; dt < 4; ++dt) ctxAcc[dt][i] = ctxAcc[dt][i] * alpha;
      int m = i + 8 * g;
      pA[w][m][ln]      = (bf16_t)p0;  // C layout -> [t][s] in LDS
      pA[w][m][16 + ln] = (bf16_t)p1;
    }
    asm volatile("s_wait_dscnt 0" ::: "memory");

    // reload probs in A layout
    V16U ap;
    ap.h[0] = *(const v8bf*)&pA[w][ln][g * 16];
    ap.h[1] = *(const v8bf*)&pA[w][ln][g * 16 + 8];

    // ---- PV: ctx += P(16x32) . V(32x64); V^T tile staged in LDS ----
#pragma unroll
    for (int dt = 0; dt < 4; ++dt) {
      const bf16_t* vr = &vTile[buf][dt * 16 + ln][g * 16];
      ctxAcc[dt] = wmma_bf16(ap.v, *(const v16bf*)vr, ctxAcc[dt]);
    }
  }

  // ---- normalize and store ctx as bf16 [row = t*B+b, col = h*64+d] ----
#pragma unroll
  for (int i = 0; i < 8; ++i) {
    float inv = 1.0f / lrow[i];
    int t = t0 + i + 8 * g;
#pragma unroll
    for (int dt = 0; dt < 4; ++dt) {
      ctx[((size_t)t * B_DIM + b) * E_DIM + h * D_DIM + dt * 16 + ln] =
          (bf16_t)(ctxAcc[dt][i] * inv);
    }
  }
}

// ---------------- host side ----------------
extern "C" void kernel_launch(void* const* d_in, const int* in_sizes, int n_in,
                              void* d_out, int out_size, void* d_ws, size_t ws_size,
                              hipStream_t stream) {
  (void)in_sizes; (void)n_in; (void)out_size; (void)ws_size;
  const float* x    = (const float*)d_in[0];
  const float* pos  = (const float*)d_in[1];
  const float* Win  = (const float*)d_in[2];
  const float* bin  = (const float*)d_in[3];
  const float* Wpos = (const float*)d_in[4];
  const float* bpos = (const float*)d_in[5];
  const float* Wout = (const float*)d_in[6];
  const float* bout = (const float*)d_in[7];
  const float* rwb  = (const float*)d_in[8];
  const float* rrb  = (const float*)d_in[9];
  float* out = (float*)d_out;

  char* cur = (char*)d_ws;
  auto alloc = [&](size_t bytes) -> char* {
    char* p = cur;
    cur += (bytes + 255) & ~(size_t)255;
    return p;
  };
  bf16_t* x_bf    = (bf16_t*)alloc((size_t)MROWS * E_DIM * 2);
  bf16_t* Win_bf  = (bf16_t*)alloc((size_t)3 * E_DIM * E_DIM * 2);
  bf16_t* Wpos_bf = (bf16_t*)alloc((size_t)E_DIM * E_DIM * 2);
  bf16_t* Wout_bf = (bf16_t*)alloc((size_t)E_DIM * E_DIM * 2);
  bf16_t* pos_bf  = (bf16_t*)alloc((size_t)R_PAD * E_DIM * 2);
  float*  qkv_f   = (float*)alloc((size_t)MROWS * 3 * E_DIM * 4);
  float*  p_f     = (float*)alloc((size_t)R_PAD * E_DIM * 4);
  bf16_t* qw_bf   = (bf16_t*)alloc((size_t)B_DIM * H_DIM * T_DIM * D_DIM * 2);
  bf16_t* qr_bf   = (bf16_t*)alloc((size_t)B_DIM * H_DIM * T_DIM * D_DIM * 2);
  bf16_t* k_bf    = (bf16_t*)alloc((size_t)B_DIM * H_DIM * T_DIM * D_DIM * 2);
  bf16_t* vT_bf   = (bf16_t*)alloc((size_t)B_DIM * H_DIM * T_DIM * D_DIM * 2);
  bf16_t* p_bf    = (bf16_t*)alloc((size_t)R_PAD * E_DIM * 2);
  bf16_t* ctx_bf  = (bf16_t*)alloc((size_t)MROWS * E_DIM * 2);

  const int thr = 256;
  cvt_f32_bf16<<<2048, thr, 0, stream>>>(x, x_bf, MROWS * E_DIM);
  cvt_f32_bf16<<<2048, thr, 0, stream>>>(Win, Win_bf, 3 * E_DIM * E_DIM);
  cvt_f32_bf16<<<2048, thr, 0, stream>>>(Wpos, Wpos_bf, E_DIM * E_DIM);
  cvt_f32_bf16<<<2048, thr, 0, stream>>>(Wout, Wout_bf, E_DIM * E_DIM);
  cvt_pad_rows<<<2048, thr, 0, stream>>>(pos, pos_bf, 2 * T_DIM - 1, E_DIM, R_PAD * E_DIM);

  gemm_bf16_wmma<<<dim3(3 * E_DIM / 64, MROWS / 64), 128, 0, stream>>>(
      qkv_f, x_bf, Win_bf, bin, MROWS, 3 * E_DIM, E_DIM);
  gemm_bf16_wmma<<<dim3(E_DIM / 64, R_PAD / 64), 128, 0, stream>>>(
      p_f, pos_bf, Wpos_bf, bpos, R_PAD, E_DIM, E_DIM);

  prep_qkv<<<8192, thr, 0, stream>>>(qkv_f, rwb, rrb, qw_bf, qr_bf, k_bf, vT_bf);
  cvt_pad_rows<<<2048, thr, 0, stream>>>(p_f, p_bf, 2 * T_DIM - 1, E_DIM, R_PAD * E_DIM);

  attn_wmma<<<dim3(T_DIM / 64, B_DIM * H_DIM), 128, 0, stream>>>(
      qw_bf, qr_bf, k_bf, vT_bf, p_bf, ctx_bf);

  gemm_bf16_wmma<<<dim3(E_DIM / 64, MROWS / 64), 128, 0, stream>>>(
      out, ctx_bf, Wout_bf, bout, MROWS, E_DIM, E_DIM);
}
